// AttentionalPoolerWMasking_83451214561333
// MI455X (gfx1250) — compile-verified, matrix operand order checked by reading, not run on hardware
//
#include <hip/hip_runtime.h>

// ---------------------------------------------------------------------------
// Types for CDNA5 WMMA (wave32, 16x16x32 bf16 -> f32) and TDM descriptors
// ---------------------------------------------------------------------------
typedef __bf16 v16bf __attribute__((ext_vector_type(16)));
typedef __bf16 v8bf  __attribute__((ext_vector_type(8)));
typedef float  v8f   __attribute__((ext_vector_type(8)));
typedef unsigned u32x4 __attribute__((ext_vector_type(4)));
typedef int      i32x4 __attribute__((ext_vector_type(4)));
typedef int      i32x8 __attribute__((ext_vector_type(8)));

#define DMODEL 768
#define CTXDIM 1024
#define NHEAD 8
#define HD 96
#define NQ 256
#define BATCH 32
#define SEQL 1024
#define KSTAGE 256   // K elements staged per TDM transfer (64 rows x 256 x 2B = 32KB)

__device__ __forceinline__ __bf16 f2bf(float f) {
  union { float f; unsigned u; } x; x.f = f;
  unsigned r = x.u + 0x7fffu + ((x.u >> 16) & 1u);   // round-to-nearest-even
  unsigned short h = (unsigned short)(r >> 16);
  return __builtin_bit_cast(__bf16, h);
}

// Per-lane fragment loader for A/B operands of v_wmma_f32_16x16x32_bf16.
// Caller passes p = base + row*ld + hg*8 (+kBase). Elements: K[hg*8 .. +7]
// and K[16+hg*8 .. +7]  (CDNA5 ISA 16-bit A/B VGPR layout).
__device__ __forceinline__ v16bf load_frag_g(const __bf16* __restrict__ p) {
  v8bf lo = *(const v8bf*)p;
  v8bf hi = *(const v8bf*)(p + 16);
  v16bf o;
#pragma unroll
  for (int i = 0; i < 8; ++i) { o[i] = lo[i]; o[i + 8] = hi[i]; }
  return o;
}

// Same, from (LDS) pointer without restrict — lowers to ds_load_b128.
__device__ __forceinline__ v16bf load_frag_l(const __bf16* p) {
  v8bf lo = *(const v8bf*)p;
  v8bf hi = *(const v8bf*)(p + 16);
  v16bf o;
#pragma unroll
  for (int i = 0; i < 8; ++i) { o[i] = lo[i]; o[i + 8] = hi[i]; }
  return o;
}

// Fragment from f32 LDS (softmax probabilities), converted to bf16.
__device__ __forceinline__ v16bf load_frag_lds_f32(const float* p) {
  v16bf o;
#pragma unroll
  for (int i = 0; i < 8; ++i) { o[i] = f2bf(p[i]); o[i + 8] = f2bf(p[16 + i]); }
  return o;
}

// ---------------------------------------------------------------------------
// Tensor Data Mover: 2D tile (tileK x rows) of bf16, global -> LDS.
// Descriptor per CDNA5 ISA (D# groups 0/1; remaining
// groups zero => dims 2..4 unused). data_size=1 (2 bytes). type=2 ("image").
// 6-arg builtin form (this toolchain): (u32x4, i32x8, i32x4, i32x4, i32x8, cpol)
// ---------------------------------------------------------------------------
__device__ __forceinline__ void tdm_load_2d_bf16(unsigned lds_off, const __bf16* gptr,
                                                 int tileK, int rows, int strideK) {
  unsigned long long ga = (unsigned long long)gptr;
  u32x4 g0;
  g0[0] = 1u;                                              // count=1, user mode
  g0[1] = lds_off;                                         // lds_addr (bytes)
  g0[2] = (unsigned)(ga & 0xffffffffu);                    // global_addr[31:0]
  g0[3] = (unsigned)((ga >> 32) & 0x01ffffffu) | (2u << 30); // addr[56:32] | type=2
  i32x8 g1;
  g1[0] = 1 << 16;                                         // data_size=1 (2B), mask=0
  g1[1] = (tileK & 0xffff) << 16;                          // tensor_dim0[15:0]
  g1[2] = (rows & 0xffff) << 16;                           // dim0 hi=0 | tensor_dim1 lo
  g1[3] = (tileK & 0xffff) << 16;                          // dim1 hi=0 | tile_dim0
  g1[4] = rows & 0xffff;                                   // tile_dim1 | tile_dim2=0
  g1[5] = strideK;                                         // tensor_dim0_stride[31:0]
  g1[6] = 0;                                               // stride hi | dim1_stride lo
  g1[7] = 0;
  i32x4 z4 = {0, 0, 0, 0};
  i32x8 z8 = {0, 0, 0, 0, 0, 0, 0, 0};
  __builtin_amdgcn_tensor_load_to_lds(g0, g1, z4, z4, z8, 0);
}

// ---------------------------------------------------------------------------
// f32 -> bf16 conversion (weights)
// ---------------------------------------------------------------------------
__global__ void cvt_bf16_kernel(const float* __restrict__ s, __bf16* __restrict__ d, int n) {
  int i = blockIdx.x * 256 + threadIdx.x;
  if (i < n) d[i] = f2bf(s[i]);
}

// ---------------------------------------------------------------------------
// LayerNorm row kernel -> bf16 output. One block (256 thr = 8 waves) per row.
// ---------------------------------------------------------------------------
__global__ __launch_bounds__(256) void ln_to_bf16_kernel(
    const float* __restrict__ src, const float* __restrict__ w,
    const float* __restrict__ bv, __bf16* __restrict__ dst, int n) {
  const int row = blockIdx.x, tid = threadIdx.x;
  const float* p = src + (size_t)row * n;
  float s = 0.f, s2 = 0.f;
  for (int c = tid; c < n; c += 256) { float v = p[c]; s += v; s2 += v * v; }
#pragma unroll
  for (int off = 16; off > 0; off >>= 1) { s += __shfl_down(s, off); s2 += __shfl_down(s2, off); }
  __shared__ float rs[8], rs2[8];
  __shared__ float mu_s, rstd_s;
  if ((tid & 31) == 0) { rs[tid >> 5] = s; rs2[tid >> 5] = s2; }
  __syncthreads();
  if (tid == 0) {
    float a = 0.f, b = 0.f;
    for (int i = 0; i < 8; ++i) { a += rs[i]; b += rs2[i]; }
    float mu = a / n;
    float var = b / n - mu * mu;
    mu_s = mu; rstd_s = rsqrtf(var + 1e-5f);
  }
  __syncthreads();
  const float mu = mu_s, rstd = rstd_s;
  __bf16* q = dst + (size_t)row * n;
  for (int c = tid; c < n; c += 256)
    q[c] = f2bf((p[c] - mu) * rstd * w[c] + bv[c]);
}

// ---------------------------------------------------------------------------
// Additive attention bias: log(size clamped) + mask, per (b,l).
// ---------------------------------------------------------------------------
__global__ void bias_kernel(const float* __restrict__ szp, const float* __restrict__ mask,
                            float* __restrict__ biasBL, int total) {
  int i = blockIdx.x * 256 + threadIdx.x;
  if (i < total) {
    float s = szp[i];
    if (s < 0.5f) s = 1.0f;
    biasBL[i] = __logf(s) + mask[i];
  }
}

// ---------------------------------------------------------------------------
// bf16 WMMA GEMM with TDM-staged B:  D[m,n] = (sum_k A[m,k]*Bw[n,k]+bias[n])*alpha
// A: [M,K] bf16 row-major (per-wave rows, streamed from global, register-
// pipelined). Bw: [N,K] bf16 row-major. The 64-row B tile used by ALL 8 waves
// of the block is DMA'd by the Tensor Data Mover into double-buffered LDS
// (KSTAGE columns per stage); wave 0 issues the next stage's tensor_load_to_lds
// before compute and publishes it with s_wait_tensorcnt + barrier.
// Block = 8 waves x (16x64 wave tile) = 128x64. Requires M%128==0, N%64==0,
// K%KSTAGE==0 (all launches satisfy this).
// mode 0: f32 row-major out   mode 1: bf16 row-major out
// mode 2: bf16 transposed-batched out: out[(b*N+n)*1024 + l], b=m>>10, l=m&1023
// ---------------------------------------------------------------------------
__global__ __launch_bounds__(256) void gemm_bf16_wmma_kernel(
    const __bf16* __restrict__ A, const __bf16* __restrict__ Bw,
    const float* __restrict__ biasv, float alpha,
    int M, int N, int K, int mode,
    __bf16* __restrict__ outBf, float* __restrict__ outF) {
  __shared__ __bf16 bs[2][64 * KSTAGE];   // 2 x 32KB double buffer for B tile

  const int tid = threadIdx.x;
  const int w = tid >> 5, lane = tid & 31, r = lane & 15, hg = lane >> 4;
  const int m0 = blockIdx.y * 128 + w * 16;
  const int n0 = blockIdx.x * 64;

  v8f acc[4] = {};
  const __bf16* ap = A + (size_t)(m0 + r) * K + hg * 8;
  const __bf16* bblk = Bw + (size_t)n0 * K;   // B tile origin (row n0, col 0)
  const int S = K / KSTAGE;

  // Prologue: DMA stage 0 into buffer 0
  if (w == 0) {
    tdm_load_2d_bf16((unsigned)(unsigned long long)&bs[0][0], bblk, KSTAGE, 64, K);
    __builtin_amdgcn_s_wait_tensorcnt(0);
  }
  __syncthreads();

  for (int s = 0; s < S; ++s) {
    const int buf = s & 1;
    // Kick off next stage's DMA into the other buffer (overlaps compute)
    if (w == 0 && s + 1 < S)
      tdm_load_2d_bf16((unsigned)(unsigned long long)&bs[buf ^ 1][0],
                       bblk + (size_t)(s + 1) * KSTAGE, KSTAGE, 64, K);

    const __bf16* bsl = &bs[buf][0] + (size_t)r * KSTAGE + hg * 8;
    const __bf16* apk = ap + s * KSTAGE;

    // Register-pipelined A stream; B fragments from LDS (ds_load_b128)
    v16bf aC = load_frag_g(apk);
#pragma unroll
    for (int c = 0; c < KSTAGE / 32; ++c) {
      v16bf aN;
      if (c < KSTAGE / 32 - 1) aN = load_frag_g(apk + (c + 1) * 32);
#pragma unroll
      for (int t = 0; t < 4; ++t) {
        v16bf bfr = load_frag_l(bsl + (size_t)t * 16 * KSTAGE + c * 32);
        acc[t] = __builtin_amdgcn_wmma_f32_16x16x32_bf16(
            false, aC, false, bfr, (short)0, acc[t], false, false);
      }
      aC = aN;
    }

    // Publish next stage: wave 0 drains TENSORcnt, then block-wide barrier.
    if (w == 0 && s + 1 < S) __builtin_amdgcn_s_wait_tensorcnt(0);
    __syncthreads();
  }

#pragma unroll
  for (int t = 0; t < 4; ++t) {
#pragma unroll
    for (int j = 0; j < 8; ++j) {
      const int m = m0 + hg * 8 + j;        // C/D layout: VGPR j -> M = j + 8*hg
      const int n = n0 + t * 16 + r;        // N = lane & 15
      const float v = (acc[t][j] + biasv[n]) * alpha;
      if (mode == 0)      outF[(size_t)m * N + n] = v;
      else if (mode == 1) outBf[(size_t)m * N + n] = f2bf(v);
      else {
        const int b = m >> 10, l = m & 1023;
        outBf[((size_t)b * N + n) * 1024 + l] = f2bf(v);
      }
    }
  }
}

// ---------------------------------------------------------------------------
// Fused attention: scores (WMMA) + bias -> LDS, softmax over L, attn@V (WMMA).
// Block = (b, h, 16-query tile). 8 waves: each does a 16x128 score slab;
// softmax is done 16-lanes-per-row with width-16 shuffles; waves 0..5 each
// compute one 16x16 d-tile of the output over K = L = 1024.
// ---------------------------------------------------------------------------
__global__ __launch_bounds__(256) void attn_kernel(
    const __bf16* __restrict__ qh,    // [NQ, 768], scale pre-baked
    const __bf16* __restrict__ kh,    // [B, 1024, 768]
    const __bf16* __restrict__ vhT,   // [B, 768, 1024]  (d-major, l contiguous)
    const float* __restrict__ biasBL, // [B, 1024]
    __bf16* __restrict__ attn_out) {  // [B, NQ, 768]
  __shared__ float sc[16 * 1024];     // 64KB: scores, then probs in place

  const int tid = threadIdx.x;
  const int w = tid >> 5, lane = tid & 31, r = lane & 15, hg = lane >> 4;
  const int bh = blockIdx.x, b = bh >> 3, h = bh & 7;
  const int q0 = blockIdx.y * 16;

  const __bf16* qbase = qh + (size_t)(q0 + r) * DMODEL + h * HD + hg * 8;
  const __bf16* kbase = kh + (size_t)b * SEQL * DMODEL + h * HD + hg * 8;
  const float* brow = biasBL + b * SEQL;

  // Phase 1: scores = (qh*scale) . kh^T + bias  -> LDS
  {
    const int lbase = w * 128;
    // A fragments (one query row, hd=96 = 3 chunks) are invariant: hoist.
    v16bf a0 = load_frag_g(qbase);
    v16bf a1 = load_frag_g(qbase + 32);
    v16bf a2 = load_frag_g(qbase + 64);
    // Prologue: K fragments for tile 0
    const __bf16* krow = kbase + (size_t)(lbase + r) * DMODEL;
    v16bf b0 = load_frag_g(krow);
    v16bf b1 = load_frag_g(krow + 32);
    v16bf b2 = load_frag_g(krow + 64);
#pragma unroll
    for (int nt = 0; nt < 8; ++nt) {
      const int l0 = lbase + nt * 16;
      v16bf n0f, n1f, n2f;
      if (nt < 7) {  // prefetch next tile's K fragments
        const __bf16* krn = kbase + (size_t)(l0 + 16 + r) * DMODEL;
        n0f = load_frag_g(krn);
        n1f = load_frag_g(krn + 32);
        n2f = load_frag_g(krn + 64);
      }
      v8f acc = {};
      acc = __builtin_amdgcn_wmma_f32_16x16x32_bf16(false, a0, false, b0, (short)0, acc, false, false);
      acc = __builtin_amdgcn_wmma_f32_16x16x32_bf16(false, a1, false, b1, (short)0, acc, false, false);
      acc = __builtin_amdgcn_wmma_f32_16x16x32_bf16(false, a2, false, b2, (short)0, acc, false, false);
      const float bl = brow[l0 + r];
#pragma unroll
      for (int j = 0; j < 8; ++j)
        sc[(hg * 8 + j) * SEQL + (l0 + r)] = acc[j] + bl;
      b0 = n0f; b1 = n1f; b2 = n2f;
    }
  }
  __syncthreads();

  // Phase 2: softmax over L (16 threads per query row; in-place exp/normalize)
  {
    const int row = tid >> 4, s = tid & 15;
    float* p = sc + row * SEQL;
    float mx = -3.4e38f;
    for (int j = s; j < SEQL; j += 16) mx = fmaxf(mx, p[j]);
#pragma unroll
    for (int off = 8; off > 0; off >>= 1) mx = fmaxf(mx, __shfl_xor(mx, off, 16));
    float sum = 0.f;
    for (int j = s; j < SEQL; j += 16) { float e = __expf(p[j] - mx); p[j] = e; sum += e; }
#pragma unroll
    for (int off = 8; off > 0; off >>= 1) sum += __shfl_xor(sum, off, 16);
    const float inv = 1.f / sum;
    for (int j = s; j < SEQL; j += 16) p[j] *= inv;
  }
  __syncthreads();

  // Phase 3: out[q, d] = probs . V   (waves 0..5 cover d = 6 x 16 = 96)
  if (w < 6) {
    const int n0 = w * 16;
    const __bf16* vbase = vhT + ((size_t)b * DMODEL + h * HD + n0 + r) * SEQL + hg * 8;
    const float* pbase = sc + r * SEQL + hg * 8;
    v8f acc = {};
    // Software-pipelined over K = L = 1024 (32 chunks)
    v16bf aC = load_frag_lds_f32(pbase);
    v16bf bC = load_frag_g(vbase);
#pragma unroll 2
    for (int kc = 32; kc < SEQL; kc += 32) {
      v16bf aN = load_frag_lds_f32(pbase + kc);
      v16bf bN = load_frag_g(vbase + kc);
      acc = __builtin_amdgcn_wmma_f32_16x16x32_bf16(
          false, aC, false, bC, (short)0, acc, false, false);
      aC = aN; bC = bN;
    }
    acc = __builtin_amdgcn_wmma_f32_16x16x32_bf16(
        false, aC, false, bC, (short)0, acc, false, false);
#pragma unroll
    for (int j = 0; j < 8; ++j)
      attn_out[((size_t)b * NQ + q0 + hg * 8 + j) * DMODEL + h * HD + n0 + r] = f2bf(acc[j]);
  }
}

// ---------------------------------------------------------------------------
// Host launcher
// ---------------------------------------------------------------------------
extern "C" void kernel_launch(void* const* d_in, const int* in_sizes, int n_in,
                              void* d_out, int out_size, void* d_ws, size_t ws_size,
                              hipStream_t stream) {
  (void)in_sizes; (void)n_in; (void)out_size; (void)ws_size;
  const float* x     = (const float*)d_in[0];
  const float* szp   = (const float*)d_in[1];
  const float* mask  = (const float*)d_in[2];
  const float* query = (const float*)d_in[3];
  const float* lnqw  = (const float*)d_in[4];
  const float* lnqb  = (const float*)d_in[5];
  const float* lnkw  = (const float*)d_in[6];
  const float* lnkb  = (const float*)d_in[7];
  const float* Wq    = (const float*)d_in[8];
  const float* Wk    = (const float*)d_in[9];
  const float* Wv    = (const float*)d_in[10];
  const float* bq    = (const float*)d_in[11];
  const float* bk    = (const float*)d_in[12];
  const float* bv    = (const float*)d_in[13];
  const float* Wo    = (const float*)d_in[14];
  const float* bo    = (const float*)d_in[15];
  float* out = (float*)d_out;

  char* ws = (char*)d_ws;
  size_t off = 0;
  auto alloc = [&](size_t bytes) -> void* {
    void* p = ws + off;
    off = (off + bytes + 255) & ~(size_t)255;
    return p;
  };
  __bf16* xk   = (__bf16*)alloc((size_t)BATCH * SEQL * CTXDIM * 2);   // 64 MB
  __bf16* khp  = (__bf16*)alloc((size_t)BATCH * SEQL * DMODEL * 2);   // 48 MB
  __bf16* vhT  = (__bf16*)alloc((size_t)BATCH * DMODEL * SEQL * 2);   // 48 MB
  __bf16* qln  = (__bf16*)alloc((size_t)NQ * DMODEL * 2);
  __bf16* qhp  = (__bf16*)alloc((size_t)NQ * DMODEL * 2);
  __bf16* aout = (__bf16*)alloc((size_t)BATCH * NQ * DMODEL * 2);     // 12 MB
  __bf16* wqb  = (__bf16*)alloc((size_t)DMODEL * DMODEL * 2);
  __bf16* wkb  = (__bf16*)alloc((size_t)DMODEL * CTXDIM * 2);
  __bf16* wvb  = (__bf16*)alloc((size_t)DMODEL * CTXDIM * 2);
  __bf16* wob  = (__bf16*)alloc((size_t)DMODEL * DMODEL * 2);
  float*  bBL  = (float*)alloc((size_t)BATCH * SEQL * 4);

  // Stage weights in bf16
  cvt_bf16_kernel<<<(DMODEL * DMODEL + 255) / 256, 256, 0, stream>>>(Wq, wqb, DMODEL * DMODEL);
  cvt_bf16_kernel<<<(DMODEL * CTXDIM + 255) / 256, 256, 0, stream>>>(Wk, wkb, DMODEL * CTXDIM);
  cvt_bf16_kernel<<<(DMODEL * CTXDIM + 255) / 256, 256, 0, stream>>>(Wv, wvb, DMODEL * CTXDIM);
  cvt_bf16_kernel<<<(DMODEL * DMODEL + 255) / 256, 256, 0, stream>>>(Wo, wob, DMODEL * DMODEL);

  // LayerNorms -> bf16
  ln_to_bf16_kernel<<<BATCH * SEQL, 256, 0, stream>>>(x, lnkw, lnkb, xk, CTXDIM);
  ln_to_bf16_kernel<<<NQ, 256, 0, stream>>>(query, lnqw, lnqb, qln, DMODEL);

  // Additive bias rows
  bias_kernel<<<(BATCH * SEQL + 255) / 256, 256, 0, stream>>>(szp, mask, bBL, BATCH * SEQL);

  const float scale = 0.1020620726159658f;  // 1/sqrt(96), baked into qh

  // qh = (LN(q) @ Wq^T + bq) * scale         [256 x 768], K=768
  gemm_bf16_wmma_kernel<<<dim3(DMODEL / 64, NQ / 128), 256, 0, stream>>>(
      qln, wqb, bq, scale, NQ, DMODEL, DMODEL, 1, qhp, nullptr);
  // kh = xk @ Wk^T + bk                      [32768 x 768], K=1024
  gemm_bf16_wmma_kernel<<<dim3(DMODEL / 64, (BATCH * SEQL) / 128), 256, 0, stream>>>(
      xk, wkb, bk, 1.0f, BATCH * SEQL, DMODEL, CTXDIM, 1, khp, nullptr);
  // vhT = (xk @ Wv^T + bv) stored [b][d][l]  (K-contiguous for attn@V)
  gemm_bf16_wmma_kernel<<<dim3(DMODEL / 64, (BATCH * SEQL) / 128), 256, 0, stream>>>(
      xk, wvb, bv, 1.0f, BATCH * SEQL, DMODEL, CTXDIM, 2, vhT, nullptr);

  // Fused scores + softmax + attn@V
  attn_kernel<<<dim3(BATCH * NHEAD, NQ / 16), 256, 0, stream>>>(qhp, khp, vhT, bBL, aout);

  // Final projection: out = aout @ Wo^T + bo  -> f32 d_out  [8192 x 768], K=768
  gemm_bf16_wmma_kernel<<<dim3(DMODEL / 64, (BATCH * NQ) / 128), 256, 0, stream>>>(
      aout, wob, bo, 1.0f, BATCH * NQ, DMODEL, DMODEL, 0, nullptr, out);
}
